// WindowMultiheadSelfAttention_58402965291628
// MI455X (gfx1250) — matrix operand order repaired and checked by reading, not compiled, not run on hardware
//
#include <hip/hip_runtime.h>
#include <hip/hip_bf16.h>
#include <cstdint>

typedef __attribute__((ext_vector_type(16))) _Float16 v16h;
typedef __attribute__((ext_vector_type(8)))  _Float16 half8;
typedef __attribute__((ext_vector_type(4)))  _Float16 half4;
typedef __attribute__((ext_vector_type(8)))  float    v8f;

union Frag16 { v16h v; half8 h[2]; };

#define XS 136   // LDS row stride (halves) for 128-wide rows; 272B, 16B aligned
#define VS 72    // LDS row stride (halves) for 64-wide rows; 144B, 16B aligned

#define NWIN 8   // windows per block (weight reuse factor)

// LDS region offsets (bytes): 4 persistent transposed f16 weights + working set
#define OFF_WQ    0
#define OFF_WK    (OFF_WQ   + 128*XS*2)    // 34816
#define OFF_WV    (OFF_WK   + 128*XS*2)    // 69632
#define OFF_WO    (OFF_WV   + 128*XS*2)    // 104448
#define OFF_XW    (OFF_WO   + 128*XS*2)    // 139264
#define OFF_Q     (OFF_XW   + 64*XS*2)     // 156672 ; reused as sO late in each window
#define OFF_K     (OFF_Q    + 64*XS*2)     // 174080
#define OFF_VT    (OFF_K    + 64*XS*2)     // 191488 ; V transposed [128][64]
#define OFF_ATTN  (OFF_VT   + 128*VS*2)    // 209920 ; attn f16 [4][64][64]
#define OFF_BIAS  (OFF_ATTN + 4*64*VS*2)   // 246784 ; rel-pos bias table 225x4 f32
#define OFF_WID   (OFF_BIAS + 225*4*4)     // 250384 ; per-token window id
#define SMEM_BYTES (OFF_WID + 64*4)        // 250640  (< 320KB WGP LDS)

// A-matrix 16x32 f16 fragment: rows row0..row0+15, K cols k0..k0+31.
// Contiguous k order (same in load_b): lane L holds row (L&15); halves[0..15]=K kb..kb+15, kb=16*(L>=16)
__device__ __forceinline__ v16h load_a(const _Float16* base, int row0, int rstride,
                                       int k0, int lane) {
  int r  = row0 + (lane & 15);
  int kb = k0 + ((lane >> 4) << 4);
  Frag16 f;
  f.h[0] = *(const half8*)(base + r * rstride + kb);
  f.h[1] = *(const half8*)(base + r * rstride + kb + 8);
  return f.v;
}

// B-matrix 32x16 f16 fragment from K-contiguous storage: baseT[n*nstride + k] = B[k][n].
// lane L holds col (L&15); K = kb..kb+15, kb=16*(L>=16)
__device__ __forceinline__ v16h load_b(const _Float16* baseT, int n0, int nstride,
                                       int k0, int lane) {
  int n  = n0 + (lane & 15);
  int kb = k0 + ((lane >> 4) << 4);
  Frag16 f;
  f.h[0] = *(const half8*)(baseT + n * nstride + kb);
  f.h[1] = *(const half8*)(baseT + n * nstride + kb + 8);
  return f.v;
}

// ---- pre-kernel: f32 weights -> f16 transposed Wt[n*128+k], 4 matrices in d_ws ----
__global__ void __launch_bounds__(256)
convert_weights_kernel(const float* __restrict__ wq, const float* __restrict__ wk,
                       const float* __restrict__ wv, const float* __restrict__ wo,
                       _Float16* __restrict__ wt) {
  int i = blockIdx.x * 256 + threadIdx.x;     // 4 * 16384 elements
  if (i >= 4 * 16384) return;
  int m = i >> 14;
  int r = i & 16383;
  int k = r >> 7, n = r & 127;
  const float* W = (m == 0) ? wq : ((m == 1) ? wk : ((m == 2) ? wv : wo));
  wt[(m << 14) + n * 128 + k] = (_Float16)W[r];
}

__global__ void __launch_bounds__(256, 1)
swin_wmsa_kernel(const float* __restrict__ x,
                 const _Float16* __restrict__ wt,     // 4x 128x128 f16, transposed
                 const float* __restrict__ bq, const float* __restrict__ bk,
                 const float* __restrict__ bv, const float* __restrict__ bo,
                 const float* __restrict__ bias_table,
                 const int*   __restrict__ shift_p,
                 float* __restrict__ out)
{
  extern __shared__ char smem[];
  _Float16* sWm[4] = { (_Float16*)(smem + OFF_WQ), (_Float16*)(smem + OFF_WK),
                       (_Float16*)(smem + OFF_WV), (_Float16*)(smem + OFF_WO) };
  _Float16* sXw   = (_Float16*)(smem + OFF_XW);
  _Float16* sQ    = (_Float16*)(smem + OFF_Q);    // reused as sO
  _Float16* sK    = (_Float16*)(smem + OFF_K);
  _Float16* sVt   = (_Float16*)(smem + OFF_VT);
  _Float16* sAt   = (_Float16*)(smem + OFF_ATTN);
  float*    sBias = (float*)   (smem + OFF_BIAS);
  int*      sWid  = (int*)     (smem + OFF_WID);

  const int tid  = threadIdx.x;
  const int lane = tid & 31;
  const int wvid = tid >> 5;            // wave 0..7

  const int shift = shift_p[0];
  const int ss    = shift ? 4 : 0;      // WS/2

  // ---- once per block: stage all 4 f16 transposed weights + bias table ----
  for (int m = 0; m < 4; ++m) {
    const _Float16* src = wt + (m << 14);
    _Float16* dst = sWm[m];
    for (int i = tid; i < 128 * 16; i += 256) {       // half8 chunks
      int n = i >> 4, c8 = (i & 15) << 3;
      *(half8*)(dst + n * XS + c8) = *(const half8*)(src + n * 128 + c8);
    }
  }
  for (int i = tid; i < 225 * 4; i += 256) sBias[i] = bias_table[i];

  const float scale = 0.1767766952966369f;   // 1/sqrt(32)

  for (int wi = 0; wi < NWIN; ++wi) {
    const int widx = blockIdx.x * NWIN + wi;   // 0..4095 = b*256 + w
    const int b  = widx >> 8;
    const int w  = widx & 255;
    const int wr = w >> 4;
    const int wc = w & 15;

    // ---------------- phase 0: gather rolled window tile ----------------
    for (int i = tid; i < 64 * 32; i += 256) {         // float4 granules
      int s = i >> 5, c4 = (i & 31) << 2;
      int y  = ((wr << 3) + (s >> 3) + ss) & 127;
      int xp = ((wc << 3) + (s & 7)  + ss) & 127;
      float4 v4 = *(const float4*)(x + ((size_t)(b * 16384 + y * 128 + xp)) * 128 + c4);
      half4 pk;
      pk[0] = (_Float16)v4.x; pk[1] = (_Float16)v4.y;
      pk[2] = (_Float16)v4.z; pk[3] = (_Float16)v4.w;
      *(half4*)(sXw + s * XS + c4) = pk;
    }
    if (tid < 64) {
      int y  = ((wr << 3) + (tid >> 3) + ss) & 127;
      int xp = ((wc << 3) + (tid & 7)  + ss) & 127;
      sWid[tid] = (y >> 3) * 16 + (xp >> 3);
    }
    __syncthreads();

    // ---------------- phase 1: Q,K,V projections ----------------
    for (int m = 0; m < 3; ++m) {
      const float* Bm = (m == 0) ? bq : ((m == 1) ? bk : bv);
      const _Float16* sW = sWm[m];
      for (int t = 0; t < 4; ++t) {
        int tile = (wvid << 2) + t;        // 0..31 over 4x8 tile grid
        int rt = tile >> 3, ct = tile & 7;
        v8f acc = {};
#pragma unroll
        for (int kk = 0; kk < 4; ++kk) {
          v16h a  = load_a(sXw, rt * 16, XS, kk * 32, lane);
          v16h bb = load_b(sW,  ct * 16, XS, kk * 32, lane);
          acc = __builtin_amdgcn_wmma_f32_16x16x32_f16(false, a, false, bb,
                                                       (short)0, acc, false, false);
        }
        int col = ct * 16 + (lane & 15);
        float bc = Bm[col];
        if (m == 2) {
          half8 pk;                         // V stored transposed, contiguous in token
#pragma unroll
          for (int v = 0; v < 8; ++v) pk[v] = (_Float16)(acc[v] + bc);
          int mrow = rt * 16 + ((lane >> 4) << 3);
          *(half8*)(sVt + col * VS + mrow) = pk;
        } else {
          _Float16* dst = (m == 0) ? sQ : sK;
#pragma unroll
          for (int v = 0; v < 8; ++v) {
            int mrow = rt * 16 + v + ((lane >> 4) << 3);
            dst[mrow * XS + col] = (_Float16)(acc[v] + bc);
          }
        }
      }
    }
    __syncthreads();

    // ------------- phase 2: scores + in-register softmax -> attn -------------
    for (int t = 0; t < 2; ++t) {
      int task = wvid + (t << 3);              // 16 (head,row-block) tasks
      int h  = task >> 2;
      int rt = task & 3;
      v8f acc[4] = {{}, {}, {}, {}};
      v16h a = load_a(sQ, rt * 16, XS, h * 32, lane);
#pragma unroll
      for (int c = 0; c < 4; ++c) {
        v16h bb = load_b(sK, c * 16, XS, h * 32, lane);   // B = K^T (row-major K)
        acc[c] = __builtin_amdgcn_wmma_f32_16x16x32_f16(false, a, false, bb,
                                                        (short)0, acc[c], false, false);
      }
      int ncol = lane & 15;
#pragma unroll
      for (int v = 0; v < 8; ++v) {
        int mrow = rt * 16 + v + ((lane >> 4) << 3);
        int mi = mrow >> 3, mj = mrow & 7;
        int widm = sWid[mrow];
        float sv[4];
        float rmax = -3.0e38f;
#pragma unroll
        for (int c = 0; c < 4; ++c) {
          int n = c * 16 + ncol;
          int idx = (mi - (n >> 3) + 7) * 15 + (mj - (n & 7) + 7);
          float mval = (shift && (widm != sWid[n])) ? -100.0f : 0.0f;
          sv[c] = acc[c][v] * scale + sBias[idx * 4 + h] + mval;
          rmax = fmaxf(rmax, sv[c]);
        }
        // row lives in one 16-lane half-wave: xor masks 1,2,4,8 stay inside it
#pragma unroll
        for (int off = 1; off < 16; off <<= 1)
          rmax = fmaxf(rmax, __shfl_xor(rmax, off, 32));
        float rsum = 0.f;
#pragma unroll
        for (int c = 0; c < 4; ++c) { sv[c] = __expf(sv[c] - rmax); rsum += sv[c]; }
#pragma unroll
        for (int off = 1; off < 16; off <<= 1)
          rsum += __shfl_xor(rsum, off, 32);
        float inv = 1.0f / rsum;
#pragma unroll
        for (int c = 0; c < 4; ++c)
          sAt[(h * 64 + mrow) * VS + c * 16 + ncol] = (_Float16)(sv[c] * inv);
      }
    }
    __syncthreads();

    // ---------------- phase 3: out = attn @ V  -> sO (reuse sQ) ----------------
    for (int t = 0; t < 2; ++t) {
      int task = wvid + (t << 3);
      int h = task >> 2, rt = task & 3;
      v8f acc[2] = {{}, {}};
#pragma unroll
      for (int kk = 0; kk < 2; ++kk) {
        v16h a = load_a(sAt + h * 64 * VS, rt * 16, VS, kk * 32, lane);
#pragma unroll
        for (int cc = 0; cc < 2; ++cc) {
          v16h bb = load_b(sVt, h * 32 + cc * 16, VS, kk * 32, lane);
          acc[cc] = __builtin_amdgcn_wmma_f32_16x16x32_f16(false, a, false, bb,
                                                           (short)0, acc[cc], false, false);
        }
      }
#pragma unroll
      for (int cc = 0; cc < 2; ++cc)
#pragma unroll
        for (int v = 0; v < 8; ++v) {
          int mrow = rt * 16 + v + ((lane >> 4) << 3);
          sQ[mrow * XS + h * 32 + cc * 16 + (lane & 15)] = (_Float16)acc[cc][v];
        }
    }
    __syncthreads();

    // ---------------- phase 4: output projection ----------------
    for (int t = 0; t < 4; ++t) {
      int tile = (wvid << 2) + t;
      int rt = tile >> 3, ct = tile & 7;
      v8f acc = {};
#pragma unroll
      for (int kk = 0; kk < 4; ++kk) {
        v16h a  = load_a(sQ,     rt * 16, XS, kk * 32, lane);   // sQ holds sO now
        v16h bb = load_b(sWm[3], ct * 16, XS, kk * 32, lane);
        acc = __builtin_amdgcn_wmma_f32_16x16x32_f16(false, a, false, bb,
                                                     (short)0, acc, false, false);
      }
      int col = ct * 16 + (lane & 15);
      float bc = bo[col];
#pragma unroll
      for (int v = 0; v < 8; ++v) {
        int mrow = rt * 16 + v + ((lane >> 4) << 3);
        out[((size_t)widx * 64 + mrow) * 128 + col] = acc[v] + bc;
      }
    }
    __syncthreads();   // protect sQ/sXw before next window's writes
  }
}

extern "C" void kernel_launch(void* const* d_in, const int* in_sizes, int n_in,
                              void* d_out, int out_size, void* d_ws, size_t ws_size,
                              hipStream_t stream) {
  (void)in_sizes; (void)n_in; (void)out_size; (void)ws_size;
  const float* x  = (const float*)d_in[0];
  const float* wq = (const float*)d_in[1];
  const float* bq = (const float*)d_in[2];
  const float* wk = (const float*)d_in[3];
  const float* bk = (const float*)d_in[4];
  const float* wv = (const float*)d_in[5];
  const float* bv = (const float*)d_in[6];
  const float* wo = (const float*)d_in[7];
  const float* bo = (const float*)d_in[8];
  const float* bt = (const float*)d_in[9];
  const int*   sh = (const int*)d_in[10];
  float* out = (float*)d_out;
  _Float16* wt = (_Float16*)d_ws;    // 4 * 16384 f16 = 128KB scratch

  hipLaunchKernelGGL(convert_weights_kernel, dim3(256), dim3(256), 0, stream,
                     wq, wk, wv, wo, wt);
  hipLaunchKernelGGL(swin_wmsa_kernel, dim3(512), dim3(256), SMEM_BYTES, stream,
                     x, wt, bq, bk, bv, bo, bt, sh, out);
}
